// StereoGNNSmallFinetune_15710990368921
// MI455X (gfx1250) — matrix operand order repaired and checked by reading, not compiled, not run on hardware
//
#include <hip/hip_runtime.h>
#include <hip/hip_bf16.h>

typedef __attribute__((ext_vector_type(16))) _Float16 v16h;
typedef __attribute__((ext_vector_type(8)))  float    v8f;

#define NN 100000
#define EE 1600000
#define GG 4096

// pack two f32 -> one dword holding two f16 (lo = x, hi = y)
__device__ inline unsigned pack2h(float x, float y) {
  union { _Float16 h[2]; unsigned u; } p;
  p.h[0] = (_Float16)x; p.h[1] = (_Float16)y;
  return p.u;
}

// ---------------- WMMA fragment loaders (CDNA5 ISA 7.12.2 layouts) ----------------
// LDS tiles are stored K-pair-packed (uint = f16 pair (K, K+1)), so each fragment
// register is a single aligned b32 LDS load.
// A: 16x32 f16. Lane: m = lane&15, g = lane>>4. VGPR v: k2 = (v&3) + 4g + 8*(v>>2).
__device__ inline v16h load_a_frag_p(const unsigned* sm, int ld2) {
  int lane = threadIdx.x & 31;
  int m = lane & 15, g = lane >> 4;
  union { v16h h; unsigned u[8]; } a;
#pragma unroll
  for (int v = 0; v < 8; ++v) {
    int k2 = (v & 3) + 4 * g + 8 * (v >> 2);
    a.u[v] = sm[m * ld2 + k2];
  }
  return a.h;
}
// B: 32x16 f16 (KxN), pair-major [k2][n]. Lane: n = lane&15 (+tile), g = lane>>4.
// VGPR v: k2 = (v&3) + 4g + 8*(v>>2).
__device__ inline v16h load_b_frag_p(const unsigned* sm, int ld2, int ncol0) {
  int lane = threadIdx.x & 31;
  int n = (lane & 15) + ncol0, g = lane >> 4;
  union { v16h h; unsigned u[8]; } b;
#pragma unroll
  for (int v = 0; v < 8; ++v) {
    int k2 = (v & 3) + 4 * g + 8 * (v >> 2);
    b.u[v] = sm[k2 * ld2 + n];
  }
  return b.h;
}
// C/D: VGPR r -> row = r + 8*(lane>>4), col = lane&15 (+tile offset)

__device__ inline unsigned flipf(float f) {
  unsigned u = __float_as_uint(f);
  return (u & 0x80000000u) ? ~u : (u | 0x80000000u);
}
__device__ inline float unflipf(unsigned u) {
  return __uint_as_float((u & 0x80000000u) ? (u & 0x7FFFFFFFu) : ~u);
}

// ---------------- Generic fused WMMA GEMM: C = epilogue(A[MxK] @ W[KxNC] + bias) ------
// MODE 0: raw   MODE 1: tanh   MODE 2: LayerNorm(g,b)+ReLU.  NC in {64,128}. 128 thr.
// K must be even (86/18/128 all are). Wave w owns column tile w (and w+4 when NC==128).
template <int NC, int MODE>
__global__ __launch_bounds__(128) void k_gemm16(
    const float* __restrict__ A, const float* __restrict__ W,
    const float* __restrict__ bias, const float* __restrict__ lng,
    const float* __restrict__ lnb, float* __restrict__ C, int M, int K) {
  const int tid = threadIdx.x, wave = tid >> 5, lane = tid & 31;
  const long brow = (long)blockIdx.x * 16;
  __shared__ unsigned sA[16 * 17];       // 16 rows x 16 K-pairs (+1 pad)
  __shared__ unsigned sB[16 * NC];       // 16 K-pairs x NC cols
  __shared__ float sC[16 * NC];
  __shared__ float sMu[16], sRs[16];
  const int KC = (K + 31) >> 5;
  v8f acc0 = {}, acc1 = {};
  for (int kc = 0; kc < KC; ++kc) {
    const int k0 = kc << 5;
#pragma unroll
    for (int i = 0; i < 2; ++i) {            // 16x16 packed A tile, 2 per thread
      int idx = tid + i * 128;
      int r = idx >> 4, k2 = idx & 15;
      long gr = brow + r; int gk = k0 + 2 * k2;
      unsigned val = 0u;
      if (gr < M && gk < K) val = pack2h(A[gr * K + gk], A[gr * K + gk + 1]);
      sA[r * 17 + k2] = val;
    }
#pragma unroll
    for (int i = 0; i < (16 * NC) / 128; ++i) {  // 16 K-pairs x NC packed W tile
      int idx = tid + i * 128;
      int k2 = idx / NC, n = idx - k2 * NC;
      int gk = k0 + 2 * k2;
      unsigned val = 0u;
      if (gk < K) val = pack2h(W[(long)gk * NC + n], W[(long)(gk + 1) * NC + n]);
      sB[k2 * NC + n] = val;
    }
    __syncthreads();
    v16h af = load_a_frag_p(sA, 17);
    v16h bf0 = load_b_frag_p(sB, NC, wave << 4);
    acc0 = __builtin_amdgcn_wmma_f32_16x16x32_f16(false, af, false, bf0,
                                                  (short)0, acc0, false, false);
    if (NC == 128) {
      v16h bf1 = load_b_frag_p(sB, NC, (wave + 4) << 4);
      acc1 = __builtin_amdgcn_wmma_f32_16x16x32_f16(false, af, false, bf1,
                                                    (short)0, acc1, false, false);
    }
    __syncthreads();
  }
  {
    int n = (lane & 15) + (wave << 4), g = lane >> 4;
#pragma unroll
    for (int r = 0; r < 8; ++r)
      sC[(r + (g << 3)) * NC + n] = acc0[r] + bias[n];
    if (NC == 128) {
      int n1 = n + 64;
#pragma unroll
      for (int r = 0; r < 8; ++r)
        sC[(r + (g << 3)) * NC + n1] = acc1[r] + bias[n1];
    }
  }
  __syncthreads();
  if (MODE == 2) {
    if (tid < 16) {
      float s = 0.f, ss = 0.f;
      for (int c = 0; c < NC; ++c) { float v = sC[tid * NC + c]; s += v; ss += v * v; }
      float mu = s / NC;
      sMu[tid] = mu;
      sRs[tid] = rsqrtf(ss / NC - mu * mu + 1e-5f);
    }
    __syncthreads();
  }
#pragma unroll
  for (int i = 0; i < (16 * NC) / 128; ++i) {
    int idx = tid + i * 128;
    int r = idx / NC, n = idx - r * NC;
    long gr = brow + r;
    if (gr < M) {
      float v = sC[r * NC + n];
      if (MODE == 1) v = tanhf(v);
      else if (MODE == 2) {
        v = (v - sMu[r]) * sRs[r] * lng[n] + lnb[n];
        v = v > 0.f ? v : 0.f;
      }
      C[gr * NC + n] = v;
    }
  }
}

// ------------- Fused GATv2 edge kernel: ee = e2@We (WMMA), m=leaky(xl[s]+xr[d]+ee),
// ------------- logit[e][h] = sum_c m*att  (16 edges per block, 128 threads) ----------
__global__ __launch_bounds__(128) void k_gat_logit(
    const float* __restrict__ e, const float* __restrict__ la,
    const float* __restrict__ We, const float* __restrict__ xl,
    const float* __restrict__ xr, const int* __restrict__ src,
    const int* __restrict__ dst, const float* __restrict__ att,
    float* __restrict__ logit) {
  const int tid = threadIdx.x, wave = tid >> 5, lane = tid & 31;
  const long base = (long)blockIdx.x * 16;
  const long Etot = (long)EE + NN;
  __shared__ unsigned sWe[32 * 128];     // 32 K-pairs x 128 cols (K=64 packed)
  __shared__ unsigned sE[16 * 33];       // 16 rows x 32 K-pairs (+1 pad)
  __shared__ float sXLR[16 * 128];
  __shared__ float sAtt[128];
  __shared__ float sLg[32];
  __shared__ int sS[16], sD[16];
  sAtt[tid] = att[tid];
  if (tid < 32) sLg[tid] = 0.f;
  if (tid < 16) {
    long ge = base + tid;
    int s = 0, d = 0;
    if (ge < EE) { s = src[ge]; d = dst[ge]; }
    else if (ge < Etot) { s = (int)(ge - EE); d = s; }
    sS[tid] = s; sD[tid] = d;
  }
  __syncthreads();
#pragma unroll
  for (int i = 0; i < 32; ++i) {              // 32x128 packed We, 32 per thread
    int idx = tid + i * 128;
    int k2 = idx >> 7, n = idx & 127;
    sWe[idx] = pack2h(We[(2 * k2) * 128 + n], We[(2 * k2 + 1) * 128 + n]);
  }
#pragma unroll
  for (int i = 0; i < 4; ++i) {               // 16x32 packed edge-feature tile
    int idx = tid + i * 128;
    int r = idx >> 5, k2 = idx & 31;
    long ge = base + r;
    int c = 2 * k2;
    unsigned val = 0u;
    if (ge < EE) val = pack2h(e[ge * 64 + c], e[ge * 64 + c + 1]);
    else if (ge < Etot) val = pack2h(la[(ge - EE) * 64 + c], la[(ge - EE) * 64 + c + 1]);
    sE[r * 33 + k2] = val;
  }
#pragma unroll
  for (int i = 0; i < 16; ++i) {              // 16x128 gathered xl[s]+xr[d]
    int idx = tid + i * 128;
    int r = idx >> 7, c = idx & 127;
    long ge = base + r;
    float v = 0.f;
    if (ge < Etot) v = xl[(long)sS[r] * 128 + c] + xr[(long)sD[r] * 128 + c];
    sXLR[r * 128 + c] = v;
  }
  __syncthreads();
  v8f acc0 = {}, acc1 = {};                    // tile wave (head0), tile wave+4 (head1)
#pragma unroll
  for (int kc = 0; kc < 2; ++kc) {
    v16h af = load_a_frag_p(sE + kc * 16, 33);
    v16h bf0 = load_b_frag_p(sWe + kc * 16 * 128, 128, wave << 4);
    acc0 = __builtin_amdgcn_wmma_f32_16x16x32_f16(false, af, false, bf0,
                                                  (short)0, acc0, false, false);
    v16h bf1 = load_b_frag_p(sWe + kc * 16 * 128, 128, (wave + 4) << 4);
    acc1 = __builtin_amdgcn_wmma_f32_16x16x32_f16(false, af, false, bf1,
                                                  (short)0, acc1, false, false);
  }
  // epilogue: leaky_relu + att weighting, butterfly-reduce over the 16-lane group
  {
    int n0 = (lane & 15) + (wave << 4);        // head 0 column
    int n1 = n0 + 64;                          // head 1 column
    int g = lane >> 4;
#pragma unroll
    for (int r = 0; r < 8; ++r) {
      int row = r + (g << 3);
      float m0 = acc0[r] + sXLR[row * 128 + n0];
      m0 = (m0 > 0.f ? m0 : 0.2f * m0) * sAtt[n0];
      float m1 = acc1[r] + sXLR[row * 128 + n1];
      m1 = (m1 > 0.f ? m1 : 0.2f * m1) * sAtt[n1];
#pragma unroll
      for (int msk = 1; msk < 16; msk <<= 1) {
        m0 += __shfl_xor(m0, msk, 32);
        m1 += __shfl_xor(m1, msk, 32);
      }
      if ((lane & 15) == 0) {
        atomicAdd(&sLg[row * 2 + 0], m0);
        atomicAdd(&sLg[row * 2 + 1], m1);
      }
    }
  }
  __syncthreads();
  if (tid < 32) {
    long ge = base + (tid >> 1);
    if (ge < Etot) logit[ge * 2 + (tid & 1)] = sLg[tid];
  }
}

// ---------------- segment softmax / aggregation via atomics ----------------
__global__ void k_deg(const int* __restrict__ dst, float* __restrict__ deg) {
  int j = blockIdx.x * 256 + threadIdx.x;
  if (j < EE) atomicAdd(&deg[dst[j]], 1.f);
}
__global__ void k_loopsum(const float* __restrict__ e, const int* __restrict__ dst,
                          float* __restrict__ la) {
  long idx = (long)blockIdx.x * 256 + threadIdx.x;
  if (idx < (long)EE * 64) {
    int j = (int)(idx >> 6), c = (int)(idx & 63);
    atomicAdd(&la[(long)dst[j] * 64 + c], e[idx]);
  }
}
__global__ void k_loopdiv(float* __restrict__ la, const float* __restrict__ deg) {
  long idx = (long)blockIdx.x * 256 + threadIdx.x;
  if (idx < (long)NN * 64) la[idx] /= fmaxf(deg[idx >> 6], 1.f);
}
__global__ void k_max(const float* __restrict__ logit, const int* __restrict__ dst,
                      unsigned* __restrict__ mx) {
  long idx = (long)blockIdx.x * 256 + threadIdx.x;
  if (idx < ((long)EE + NN) * 2) {
    long ge = idx >> 1; int h = (int)(idx & 1);
    int d = (ge < EE) ? dst[ge] : (int)(ge - EE);
    atomicMax(&mx[d * 2 + h], flipf(logit[idx]));
  }
}
__global__ void k_mxf(const unsigned* __restrict__ mx, float* __restrict__ mxf) {
  int i = blockIdx.x * 256 + threadIdx.x;
  if (i < NN * 2) mxf[i] = unflipf(mx[i]);
}
__global__ void k_den(const float* __restrict__ logit, const int* __restrict__ dst,
                      const float* __restrict__ mxf, float* __restrict__ den) {
  long idx = (long)blockIdx.x * 256 + threadIdx.x;
  if (idx < ((long)EE + NN) * 2) {
    long ge = idx >> 1; int h = (int)(idx & 1);
    int d = (ge < EE) ? dst[ge] : (int)(ge - EE);
    atomicAdd(&den[d * 2 + h], expf(logit[idx] - mxf[d * 2 + h]));
  }
}
__global__ void k_invden(float* __restrict__ den) {
  int i = blockIdx.x * 256 + threadIdx.x;
  if (i < NN * 2) den[i] = 1.f / den[i];
}
__global__ void k_agg(const float* __restrict__ xl, const float* __restrict__ logit,
                      const float* __restrict__ mxf, const float* __restrict__ invd,
                      const int* __restrict__ src, const int* __restrict__ dst,
                      float* __restrict__ agg) {
  long ge = blockIdx.x;
  int c = threadIdx.x;
  int s, d;
  if (ge < EE) { s = src[ge]; d = dst[ge]; }
  else { s = (int)(ge - EE); d = s; }
  int h = c >> 6;
  float alpha = expf(logit[ge * 2 + h] - mxf[d * 2 + h]) * invd[d * 2 + h];
  atomicAdd(&agg[(long)d * 128 + c], xl[(long)s * 128 + c] * alpha);
}
__global__ void k_post(float* __restrict__ h, const float* __restrict__ agg,
                       const float* __restrict__ bias, const float* __restrict__ lng,
                       const float* __restrict__ lnb) {
  long n = blockIdx.x; int c = threadIdx.x;
  __shared__ float s1[128], s2[128];
  float v = agg[n * 128 + c] + bias[c];
  s1[c] = v; s2[c] = v * v; __syncthreads();
  for (int off = 64; off > 0; off >>= 1) {
    if (c < off) { s1[c] += s1[c + off]; s2[c] += s2[c + off]; }
    __syncthreads();
  }
  float mu = s1[0] * (1.f / 128.f);
  float rs = rsqrtf(s2[0] * (1.f / 128.f) - mu * mu + 1e-5f);
  float x = (v - mu) * rs * lng[c] + lnb[c];
  x = x > 0.f ? x : 0.f;
  h[n * 128 + c] = x + h[n * 128 + c];
}

// ---------------- pooling / readout / heads ----------------
__global__ void k_pool(const float* __restrict__ h, const int* __restrict__ batch,
                       float* __restrict__ pooled, float* __restrict__ cnt) {
  long n = blockIdx.x; int c = threadIdx.x;
  int b = batch[n];
  atomicAdd(&pooled[(long)b * 128 + c], h[n * 128 + c]);
  if (c == 0) atomicAdd(&cnt[b], 1.f);
}
__global__ void k_pooldiv(float* __restrict__ pooled, const float* __restrict__ cnt) {
  int i = blockIdx.x * 256 + threadIdx.x;
  if (i < GG * 128) pooled[i] /= fmaxf(cnt[i >> 7], 1.f);
}
__global__ void k_head1(const float* __restrict__ g, const float* __restrict__ W1,
                        const float* __restrict__ b1, const float* __restrict__ g1,
                        const float* __restrict__ be1, float* __restrict__ z1) {
  int row = blockIdx.x, t = threadIdx.x;
  __shared__ float s1[128], s2[128];
  float z = 0.f;
  if (t < 96) {
    const float* gr = g + (long)row * 128;
    for (int k = 0; k < 128; ++k) z += gr[k] * W1[k * 96 + t];
    z += b1[t];
  }
  s1[t] = (t < 96) ? z : 0.f; s2[t] = (t < 96) ? z * z : 0.f; __syncthreads();
  for (int off = 64; off > 0; off >>= 1) {
    if (t < off) { s1[t] += s1[t + off]; s2[t] += s2[t + off]; }
    __syncthreads();
  }
  float mu = s1[0] / 96.f;
  float rs = rsqrtf(s2[0] / 96.f - mu * mu + 1e-5f);
  if (t < 96) {
    float v = (z - mu) * rs * g1[t] + be1[t];
    z1[(long)row * 96 + t] = v > 0.f ? v : 0.f;
  }
}
__global__ void k_head2(const float* __restrict__ z1, const float* __restrict__ W2,
                        const float* __restrict__ b2, float* __restrict__ z2) {
  int row = blockIdx.x, t = threadIdx.x;
  if (t < 48) {
    float z = b2[t];
    const float* zr = z1 + (long)row * 96;
    for (int k = 0; k < 96; ++k) z += zr[k] * W2[k * 48 + t];
    z2[(long)row * 48 + t] = z > 0.f ? z : 0.f;
  }
}
__global__ void k_head3(const float* __restrict__ z2, const float* __restrict__ W3,
                        const float* __restrict__ b3, float* __restrict__ out) {
  int idx = blockIdx.x * 128 + threadIdx.x;
  if (idx < GG * 3) {
    int row = idx / 3, j = idx - row * 3;
    float z = b3[j];
    const float* zr = z2 + (long)row * 48;
    for (int k = 0; k < 48; ++k) z += zr[k] * W3[k * 3 + j];
    out[idx] = z;
  }
}

extern "C" void kernel_launch(void* const* d_in, const int* in_sizes, int n_in,
                              void* d_out, int out_size, void* d_ws, size_t ws_size,
                              hipStream_t stream) {
  (void)in_sizes; (void)n_in; (void)out_size; (void)ws_size;
  // ---- input map: JAX pytree flattening (sorted dict keys) of setup_inputs() ----
  const float* x       = (const float*)d_in[0];    // (N,86)
  const float* eattr   = (const float*)d_in[1];    // (E,18)
  const float* edge_W  = (const float*)d_in[2];    // (18,64)
  const float* edge_b  = (const float*)d_in[3];
  const float* edge_be = (const float*)d_in[4];
  const float* edge_g  = (const float*)d_in[5];
  struct Head { const float *W1, *W2, *W3, *b1, *b2, *b3, *be1, *g1; };
  Head hd[3];
  for (int t = 0; t < 3; ++t) {
    int b = 6 + t * 8;
    hd[t].W1  = (const float*)d_in[b + 0]; hd[t].W2 = (const float*)d_in[b + 1];
    hd[t].W3  = (const float*)d_in[b + 2]; hd[t].b1 = (const float*)d_in[b + 3];
    hd[t].b2  = (const float*)d_in[b + 4]; hd[t].b3 = (const float*)d_in[b + 5];
    hd[t].be1 = (const float*)d_in[b + 6]; hd[t].g1 = (const float*)d_in[b + 7];
  }
  struct Layer { const float *We, *Wl, *Wr, *att, *bias, *bl, *br, *ln_b, *ln_g; };
  Layer ly[2];
  for (int l = 0; l < 2; ++l) {
    int b = 30 + l * 9;
    ly[l].We  = (const float*)d_in[b + 0]; ly[l].Wl   = (const float*)d_in[b + 1];
    ly[l].Wr  = (const float*)d_in[b + 2]; ly[l].att  = (const float*)d_in[b + 3];
    ly[l].bias= (const float*)d_in[b + 4]; ly[l].bl   = (const float*)d_in[b + 5];
    ly[l].br  = (const float*)d_in[b + 6]; ly[l].ln_b = (const float*)d_in[b + 7];
    ly[l].ln_g= (const float*)d_in[b + 8];
  }
  const float* node_W    = (const float*)d_in[48];
  const float* node_b    = (const float*)d_in[49];
  const float* node_be   = (const float*)d_in[50];
  const float* node_g    = (const float*)d_in[51];
  const float* readout_W = (const float*)d_in[52];
  const float* readout_b = (const float*)d_in[53];
  const int* eidx  = (const int*)d_in[54];
  const int* src   = eidx;
  const int* dst   = eidx + EE;
  const int* batch = (const int*)d_in[55];
  float* out = (float*)d_out;

  // ---- workspace layout (f32) ----
  float* W = (float*)d_ws;
  float* h      = W;                    W += (long)NN * 128;
  float* e      = W;                    W += (long)EE * 64;
  float* la     = W;                    W += (long)NN * 64;
  float* deg    = W;                    W += NN;
  float* xl     = W;                    W += (long)NN * 128;
  float* xr     = W;                    W += (long)NN * 128;
  float* logit  = W;                    W += ((long)EE + NN) * 2;
  unsigned* mx  = (unsigned*)W;         W += NN * 2;
  float* mxf    = W;                    W += NN * 2;
  float* den    = W;                    W += NN * 2;
  float* pooled = W;                    W += (long)GG * 128;
  float* cnt    = W;                    W += GG;
  float* gv     = W;                    W += (long)GG * 128;
  float* z1     = W;                    W += (long)GG * 96;
  float* z2     = W;                    W += (long)GG * 48;
  float* agg    = W;                    W += (long)NN * 128;

  const long Etot = (long)EE + NN;
  // ---- encoders ----
  k_gemm16<128, 2><<<(NN + 15) / 16, 128, 0, stream>>>(x, node_W, node_b, node_g,
                                                       node_be, h, NN, 86);
  k_gemm16<64, 2><<<(EE + 15) / 16, 128, 0, stream>>>(eattr, edge_W, edge_b, edge_g,
                                                      edge_be, e, EE, 18);
  // ---- self-loop fill: deg & mean incoming edge_attr ----
  hipMemsetAsync(deg, 0, NN * sizeof(float), stream);
  hipMemsetAsync(la, 0, (long)NN * 64 * sizeof(float), stream);
  k_deg<<<(EE + 255) / 256, 256, 0, stream>>>(dst, deg);
  k_loopsum<<<(int)(((long)EE * 64 + 255) / 256), 256, 0, stream>>>(e, dst, la);
  k_loopdiv<<<(int)(((long)NN * 64 + 255) / 256), 256, 0, stream>>>(la, deg);
  // ---- GATv2 layers ----
  for (int l = 0; l < 2; ++l) {
    k_gemm16<128, 0><<<(NN + 15) / 16, 128, 0, stream>>>(h, ly[l].Wl, ly[l].bl, nullptr,
                                                         nullptr, xl, NN, 128);
    k_gemm16<128, 0><<<(NN + 15) / 16, 128, 0, stream>>>(h, ly[l].Wr, ly[l].br, nullptr,
                                                         nullptr, xr, NN, 128);
    k_gat_logit<<<(int)((Etot + 15) / 16), 128, 0, stream>>>(e, la, ly[l].We, xl, xr,
                                                             src, dst, ly[l].att, logit);
    hipMemsetAsync(mx, 0, NN * 2 * sizeof(unsigned), stream);
    hipMemsetAsync(den, 0, NN * 2 * sizeof(float), stream);
    k_max<<<(int)((Etot * 2 + 255) / 256), 256, 0, stream>>>(logit, dst, mx);
    k_mxf<<<(NN * 2 + 255) / 256, 256, 0, stream>>>(mx, mxf);
    k_den<<<(int)((Etot * 2 + 255) / 256), 256, 0, stream>>>(logit, dst, mxf, den);
    k_invden<<<(NN * 2 + 255) / 256, 256, 0, stream>>>(den);
    hipMemsetAsync(agg, 0, (long)NN * 128 * sizeof(float), stream);
    k_agg<<<(int)Etot, 128, 0, stream>>>(xl, logit, mxf, den, src, dst, agg);
    k_post<<<NN, 128, 0, stream>>>(h, agg, ly[l].bias, ly[l].ln_g, ly[l].ln_b);
  }
  // ---- pooling, readout, heads ----
  hipMemsetAsync(pooled, 0, (long)GG * 128 * sizeof(float), stream);
  hipMemsetAsync(cnt, 0, GG * sizeof(float), stream);
  k_pool<<<NN, 128, 0, stream>>>(h, batch, pooled, cnt);
  k_pooldiv<<<(GG * 128 + 255) / 256, 256, 0, stream>>>(pooled, cnt);
  k_gemm16<128, 1><<<GG / 16, 128, 0, stream>>>(pooled, readout_W, readout_b, nullptr,
                                                nullptr, gv, GG, 128);
  for (int t = 0; t < 3; ++t) {
    k_head1<<<GG, 128, 0, stream>>>(gv, hd[t].W1, hd[t].b1, hd[t].g1, hd[t].be1, z1);
    k_head2<<<GG, 64, 0, stream>>>(z1, hd[t].W2, hd[t].b2, z2);
    k_head3<<<(GG * 3 + 127) / 128, 128, 0, stream>>>(z2, hd[t].W3, hd[t].b3,
                                                      out + (long)t * GG * 3);
  }
}
